// ByteJEPA_42683384987934
// MI455X (gfx1250) — compile-verified, matrix operand order checked by reading, not run on hardware
//
#include <hip/hip_runtime.h>
#include <hip/hip_bf16.h>
#include <math.h>

typedef __attribute__((ext_vector_type(16))) _Float16 v16h;
typedef __attribute__((ext_vector_type(8)))  float    v8f;

#define BB 8
#define CC 2048
#define TT 1024
#define SS (CC + TT)          // 3072
#define NT (BB * TT)          // 8192 target elements
#define CHUNK 128
#define NCHUNK (SS / CHUNK)   // 24 j-chunks
#define TBLOCKS (NT / 256)    // 32 target tiles

// Workspace layout: int counts[NT][8]; k=0..3 -> total_n (n=1..4), k=4..7 -> true_n.

__global__ void ngram_zero(int* __restrict__ counts) {
  int i = blockIdx.x * blockDim.x + threadIdx.x;
  if (i < NT * 8) counts[i] = 0;
}

// One thread per target t; each block handles a 128-wide j-chunk for 256 targets.
// Suffix match of order n at (p, j)  <=>  top n bytes of (key ^ w(j)) are zero,
// where key packs seq[p-4..p-1] (p-1 in MSB) and w(j) packs seq[j-4..j-1].
// The sequence is staged byte-packed in LDS; the inner loop consumes one LDS
// dword per 4 positions and derives the 4 rolling windows by shifts.
__global__ void ngram_count(const int* __restrict__ ctx,
                            const int* __restrict__ tgt,
                            int* __restrict__ counts) {
  const int tt    = blockIdx.x % TBLOCKS;
  const int chunk = blockIdx.x / TBLOCKS;
  const int b    = tt >> 2;
  const int tile = tt & 3;

  __shared__ unsigned int s32[SS / 4];
  const unsigned char* s = (const unsigned char*)s32;

  for (int i = threadIdx.x; i < SS / 4; i += blockDim.x) {
    // dword i covers ids 4i..4i+3; CC is dword-aligned so each dword is
    // entirely context or entirely target.
    const int base = i * 4;
    const int* src = (base < CC) ? (ctx + b * CC + base) : (tgt + b * TT + (base - CC));
    const unsigned int v0 = (unsigned)src[0] & 0xFFu;
    const unsigned int v1 = (unsigned)src[1] & 0xFFu;
    const unsigned int v2 = (unsigned)src[2] & 0xFFu;
    const unsigned int v3 = (unsigned)src[3] & 0xFFu;
    s32[i] = v0 | (v1 << 8) | (v2 << 16) | (v3 << 24);
  }
  __syncthreads();

  const int t = tile * 256 + threadIdx.x;
  const int p = CC + t;
  const unsigned int key = (unsigned)s[p - 4]        | ((unsigned)s[p - 3] << 8) |
                           ((unsigned)s[p - 2] << 16) | ((unsigned)s[p - 1] << 24);
  const int tgtb = s[p];

  int c1 = 0, c2 = 0, c3 = 0, c4 = 0;
  int r1 = 0, r2 = 0, r3 = 0, r4 = 0;

  const int jlo = chunk * CHUNK;
  int jstart = jlo;
  if (chunk == 0) {
    // j = 1..3: only orders n <= j are valid (gram must fit: j >= n).
    const unsigned int d0 = s32[0];
    const int s0 = (int)(d0 & 0xFFu);
    const int s1 = (int)((d0 >> 8) & 0xFFu);
    const int s2 = (int)((d0 >> 16) & 0xFFu);
    const int s3 = (int)((d0 >> 24) & 0xFFu);
    const int pb1 = (int)(key >> 24);            // s[p-1]
    const int pb2 = (int)((key >> 16) & 0xFFu);  // s[p-2]
    const int pb3 = (int)((key >> 8)  & 0xFFu);  // s[p-3]
    {
      int m1 = (s0 == pb1);
      int tm = (s1 == tgtb);
      c1 += m1; r1 += m1 & tm;
    }
    {
      int m1 = (s1 == pb1);
      int m2 = m1 & (s0 == pb2);
      int tm = (s2 == tgtb);
      c1 += m1; r1 += m1 & tm;
      c2 += m2; r2 += m2 & tm;
    }
    {
      int m1 = (s2 == pb1);
      int m2 = m1 & (s1 == pb2);
      int m3 = m2 & (s0 == pb3);
      int tm = (s3 == tgtb);
      c1 += m1; r1 += m1 & tm;
      c2 += m2; r2 += m2 & tm;
      c3 += m3; r3 += m3 & tm;
    }
    jstart = 4;
  }
  const int jhi = jlo + CHUNK;

  // w packs s[jstart-4..jstart-1]; jstart is a multiple of 4 so this is the
  // aligned dword just below the chunk.
  unsigned int w = s32[(jstart >> 2) - 1];
  for (int j4 = jstart; j4 < jhi; j4 += 4) {
    const unsigned int d = s32[j4 >> 2];  // bytes s[j4..j4+3], s[j4] in LSB
#pragma unroll
    for (int k = 0; k < 4; ++k) {
      const int j = j4 + k;
      const unsigned int wk = (k == 0) ? w : ((w >> (8 * k)) | (d << (32 - 8 * k)));
      const unsigned int x = key ^ wk;
      const int bj  = (int)((d >> (8 * k)) & 0xFFu);
      const int act = (j < p);
      const int m1 = (int)((x & 0xFF000000u) == 0u) & act;
      const int m2 = (int)((x & 0xFFFF0000u) == 0u) & act;
      const int m3 = (int)((x & 0xFFFFFF00u) == 0u) & act;
      const int m4 = (int)(x == 0u) & act;
      const int tm = (bj == tgtb);
      c1 += m1; r1 += m1 & tm;
      c2 += m2; r2 += m2 & tm;
      c3 += m3; r3 += m3 & tm;
      c4 += m4; r4 += m4 & tm;
    }
    w = d;
  }

  int* dst = counts + (b * TT + t) * 8;
  atomicAdd(dst + 0, c1); atomicAdd(dst + 1, c2);
  atomicAdd(dst + 2, c3); atomicAdd(dst + 3, c4);
  atomicAdd(dst + 4, r1); atomicAdd(dst + 5, r2);
  atomicAdd(dst + 6, r3); atomicAdd(dst + 7, r4);
}

// One wave32: gate+blend each element, then WMMA-sum all 8192 values.
// A = ones => sum(D) = 16 * sum(all B entries); each f32 value is split into
// (hi,lo) f16 pair so the f16 operands carry ~f32 accuracy into the f32 acc.
__global__ void ngram_blend_reduce(const float* __restrict__ lp,
                                   const int* __restrict__ counts,
                                   float* __restrict__ out) {
  const int lane = threadIdx.x;  // 0..31, one wave, EXEC all ones
  v16h ones;
#pragma unroll
  for (int k = 0; k < 16; ++k) ones[k] = (_Float16)1.0f;
  v8f acc = {0.f, 0.f, 0.f, 0.f, 0.f, 0.f, 0.f, 0.f};

  for (int iter = 0; iter < NT / 256; ++iter) {  // 32 iters x 256 values
    v16h bm;
#pragma unroll
    for (int e = 0; e < 8; ++e) {
      const int idx = iter * 256 + lane * 8 + e;
      const int* cc = counts + idx * 8;
      int wti = 0, wtri = 0;
#pragma unroll
      for (int n = 1; n <= 4; ++n) {
        const int cn = cc[n - 1];
        const int tn = cc[4 + n - 1];
        const int ok = (cn >= 2);
        wti  += ok ? n * cn : 0;   // weight = n - MIN_N + 1 = n
        wtri += ok ? n * tn : 0;
      }
      const float wtT = (float)wti;
      const float wtR = (float)wtri;
      const float l  = lp[idx];
      const float mp = expf(l);
      const float cp = (wtR + 0.25f) / (wtT + 0.25f * 256.0f);
      const float ae = 0.3f * wtT / (wtT + 20.0f);
      const float mixed = (1.0f - ae) * mp + ae * cp;
      const float v = (wti > 0) ? -logf(fmaxf(mixed, 1e-12f)) : -l;
      const _Float16 h1 = (_Float16)v;
      const _Float16 h2 = (_Float16)(v - (float)h1);
      bm[2 * e]     = h1;
      bm[2 * e + 1] = h2;
    }
    acc = __builtin_amdgcn_wmma_f32_16x16x32_f16(
        false, ones, false, bm, (short)0, acc, false, false);
  }

  float ssum = acc[0] + acc[1] + acc[2] + acc[3] +
               acc[4] + acc[5] + acc[6] + acc[7];
#pragma unroll
  for (int off = 16; off > 0; off >>= 1)
    ssum += __shfl_xor(ssum, off, 32);
  if (lane == 0) out[0] = ssum * (1.0f / (16.0f * (float)NT));
}

extern "C" void kernel_launch(void* const* d_in, const int* in_sizes, int n_in,
                              void* d_out, int out_size, void* d_ws, size_t ws_size,
                              hipStream_t stream) {
  const float* lp  = (const float*)d_in[0];   // model_true_log_probs [B,T] f32
  const int*   ctx = (const int*)d_in[1];     // context_ids [B,C] i32
  const int*   tgt = (const int*)d_in[2];     // target_ids  [B,T] i32
  float* out   = (float*)d_out;               // scalar mean loss
  int* counts  = (int*)d_ws;                  // NT*8 ints = 256 KB scratch

  ngram_zero<<<dim3((NT * 8 + 255) / 256), dim3(256), 0, stream>>>(counts);
  ngram_count<<<dim3(TBLOCKS * NCHUNK), dim3(256), 0, stream>>>(ctx, tgt, counts);
  ngram_blend_reduce<<<dim3(1), dim3(32), 0, stream>>>(lp, counts, out);
}
